// TVNet_83339545412100
// MI455X (gfx1250) — matrix operand order repaired and checked by reading, not compiled, not run on hardware
//
#include <hip/hip_runtime.h>

#define Wd 256
#define Hd 256
#define HW 65536
#define NB 8
#define PLANE 524288            // NB*HW floats = 2MB
#define EPSF 1e-12f
#define LT   0.045f             // LBDA*THETA
#define TAUT 0.8333333f         // TAU/THETA
#define THETAv 0.3f

typedef _Float16 v16h __attribute__((ext_vector_type(16)));
typedef float    v8f  __attribute__((ext_vector_type(8)));
typedef unsigned int u32x4 __attribute__((ext_vector_type(4)));
typedef int          i32x4 __attribute__((ext_vector_type(4)));
typedef int          i32x8 __attribute__((ext_vector_type(8)));
typedef __attribute__((address_space(1))) int glb_i32_t;
typedef __attribute__((address_space(3))) int lds_i32_t;
typedef __attribute__((address_space(3))) float lds_float_t;

#if __has_builtin(__builtin_amdgcn_global_load_async_to_lds_b32) && \
    __has_builtin(__builtin_amdgcn_s_wait_asynccnt)
#define HAS_ASYNC_LDS 1
#else
#define HAS_ASYNC_LDS 0
#endif
#if __has_builtin(__builtin_amdgcn_tensor_load_to_lds) && \
    __has_builtin(__builtin_amdgcn_s_wait_tensorcnt)
#define HAS_TDM 1
#else
#define HAS_TDM 0
#endif

// ---- order-preserving float<->uint keys for atomic min/max ----
__device__ __forceinline__ unsigned fkey(float f) {
  unsigned u = __float_as_uint(f);
  return (u & 0x80000000u) ? ~u : (u | 0x80000000u);
}
__device__ __forceinline__ float funkey(unsigned k) {
  return (k & 0x80000000u) ? __uint_as_float(k ^ 0x80000000u) : __uint_as_float(~k);
}

__global__ void k_init(unsigned* mm) {
  if (threadIdx.x == 0) { mm[0] = 0xFFFFFFFFu; mm[1] = 0u; }
}

// gray conversion + global min/max over both images
__global__ void __launch_bounds__(256) k_gray(const float* __restrict__ x1,
                                              const float* __restrict__ x2,
                                              float* __restrict__ g1,
                                              float* __restrict__ g2,
                                              unsigned* __restrict__ mm) {
  int idx = blockIdx.x * 256 + threadIdx.x;          // < PLANE
  int b = idx >> 16, p = idx & 0xFFFF;
  const float* a1 = x1 + (size_t)b * 3 * HW + p;
  const float* a2 = x2 + (size_t)b * 3 * HW + p;
  float v1 = 0.114f * a1[0] + 0.587f * a1[HW] + 0.299f * a1[2 * HW];
  float v2 = 0.114f * a2[0] + 0.587f * a2[HW] + 0.299f * a2[2 * HW];
  g1[idx] = v1; g2[idx] = v2;
  __shared__ float smn[256], smx[256];
  int t = threadIdx.x;
  smn[t] = fminf(v1, v2); smx[t] = fmaxf(v1, v2);
  __syncthreads();
  for (int s = 128; s > 0; s >>= 1) {
    if (t < s) { smn[t] = fminf(smn[t], smn[t + s]); smx[t] = fmaxf(smx[t], smx[t + s]); }
    __syncthreads();
  }
  if (t == 0) { atomicMin(&mm[0], fkey(smn[0])); atomicMax(&mm[1], fkey(smx[0])); }
}

// separable Gaussian taps, branchless (avoid divergent exec-mask ladders)
__device__ __forceinline__ float wtap(int t) {
  return 0.0295608f * (float)((t == 0) | (t == 4))
       + 0.2360131f * (float)((t == 1) | (t == 3))
       + 0.4688635f * (float)(t == 2);
}

// horizontal pass: Out(16x16) = In(16x20, normalized, zero-pad) x Wband(20x16), one WMMA
__global__ void __launch_bounds__(32) k_gauss_row(const float* __restrict__ g,
                                                  float* __restrict__ t,
                                                  const unsigned* __restrict__ mm) {
  int lane = threadIdx.x;
  int tile = blockIdx.x;                // 16 planes * 256 tiles
  int img = tile >> 8;
  int ty = (tile >> 4) & 15, tx = tile & 15;
  int l = lane & 15, half = lane >> 4;
  float mn = funkey(mm[0]);
  float sc = 255.0f / (funkey(mm[1]) - mn);
  const float* gp = g + (size_t)img * HW + (ty * 16 + l) * Wd;
  v16h a, b;
  #pragma unroll
  for (int e = 0; e < 16; ++e) {
    int K = e + (e < 8 ? 0 : 8) + (half ? 8 : 0);    // A element -> K per ISA layout
    int c = tx * 16 - 2 + K;
    int cc = min(max(c, 0), Wd - 1);
    float v = (gp[cc] - mn) * sc;
    float av = (K < 20 && c >= 0 && c < Wd) ? v : 0.0f;   // zero 'SAME' pad
    a[e] = (_Float16)av;
    int Kb = e + (e < 8 ? 0 : 8) + (half ? 8 : 0);    // B element -> K, same table order as A
    b[e] = (_Float16)((Kb < 20) ? wtap(Kb - l) : 0.0f);
  }
  v8f acc = {0.f, 0.f, 0.f, 0.f, 0.f, 0.f, 0.f, 0.f};
  acc = __builtin_amdgcn_wmma_f32_16x16x32_f16(false, a, false, b, (short)0, acc, false, false);
  float* tp = t + (size_t)img * HW;
  #pragma unroll
  for (int i = 0; i < 8; ++i)
    tp[(ty * 16 + (half ? 8 : 0) + i) * Wd + tx * 16 + l] = acc[i];
}

// vertical pass: TDM DMA's the 20x16 input slab into LDS, then one WMMA:
// Out(16x16) = Wband(16x20) x In(20x16, zero-pad)
__global__ void __launch_bounds__(32) k_gauss_col(const float* __restrict__ t,
                                                  float* __restrict__ s) {
  __shared__ float lt[20][16];
  int lane = threadIdx.x;
  int tile = blockIdx.x;
  int img = tile >> 8;
  int ty = (tile >> 4) & 15, tx = tile & 15;
  int l = lane & 15, half = lane >> 4;
  const float* tp = t + (size_t)img * HW;
#if HAS_TDM
  {
    // D# group0: count=1, lds_addr, global tile addr (57b), type=2
    unsigned lds_addr = (unsigned)(size_t)(lds_float_t*)&lt[0][0];
    const float* gsrc = tp + ((ty * 16 - 2) * Wd + tx * 16); // +/-2 rows stays inside ws
    unsigned long long ga = (unsigned long long)(size_t)gsrc;
    u32x4 g0;
    g0[0] = 1u;                                       // count=1 (valid), user mode
    g0[1] = lds_addr;
    g0[2] = (unsigned)(ga & 0xFFFFFFFFu);
    g0[3] = (unsigned)((ga >> 32) & 0x1FFFFFFu) | (2u << 30);  // type=2 ("image")
    // D# group1: data_size=4B code 2; tensor_dim0/1 huge (no clip); tile 16x20; stride0=256
    i32x8 g1;
    g1[0] = 0x00020000;                               // data_size=2 (4 bytes)
    g1[1] = (int)(((1u << 20) & 0xFFFFu) << 16);      // tensor_dim0 lo16 -> [31:16]
    g1[2] = (int)(((1u << 20) >> 16) | ((((1u << 20)) & 0xFFFFu) << 16)); // dim0 hi | dim1 lo
    g1[3] = (int)(((1u << 20) >> 16) | (16u << 16));  // dim1 hi | tile_dim0=16
    g1[4] = 20;                                       // tile_dim1=20, tile_dim2=0
    g1[5] = 256;                                      // tensor_dim0_stride lo32 = 256
    g1[6] = 0;                                        // stride0 hi | stride1 lo
    g1[7] = 0;
    i32x4 z4 = {0, 0, 0, 0};
#if __clang_major__ >= 23
    i32x8 z8 = {0, 0, 0, 0, 0, 0, 0, 0};
    __builtin_amdgcn_tensor_load_to_lds(g0, g1, z4, z4, z8, 0);
#else
    __builtin_amdgcn_tensor_load_to_lds(g0, g1, z4, z4, 0);
#endif
    __builtin_amdgcn_s_wait_tensorcnt(0);             // single wave per WG: no barrier needed
  }
#else
  for (int e = lane; e < 320; e += 32) {
    int r = e >> 4, c = e & 15;
    int gy2 = ty * 16 - 2 + r;
    int cy = min(max(gy2, 0), Hd - 1);
    lt[r][c] = tp[cy * Wd + tx * 16 + c];
  }
  __syncthreads();
#endif
  v16h a, b;
  #pragma unroll
  for (int e = 0; e < 16; ++e) {
    int K = e + (e < 8 ? 0 : 8) + (half ? 8 : 0);
    a[e] = (_Float16)((K < 20) ? wtap(K - l) : 0.0f);
    int Kb = e + (e < 8 ? 0 : 8) + (half ? 8 : 0);
    float bv = 0.0f;
    if (Kb < 20) {
      int r = ty * 16 - 2 + Kb;
      bv = (r >= 0 && r < Hd) ? lt[Kb][l] : 0.0f;     // mask DMA'd halo rows outside image
    }
    b[e] = (_Float16)bv;
  }
  v8f acc = {0.f, 0.f, 0.f, 0.f, 0.f, 0.f, 0.f, 0.f};
  acc = __builtin_amdgcn_wmma_f32_16x16x32_f16(false, a, false, b, (short)0, acc, false, false);
  float* sp = s + (size_t)img * HW;
  #pragma unroll
  for (int i = 0; i < 8; ++i)
    sp[(ty * 16 + (half ? 8 : 0) + i) * Wd + tx * 16 + l] = acc[i];
}

// centered gradient of s2 (+grad, rho_c since zero-flow warp is identity) + zero-init state
__global__ void __launch_bounds__(256) k_grad(const float* __restrict__ s1,
                                              const float* __restrict__ s2,
                                              float* __restrict__ fix,   // dx,dy,grad,rhoc
                                              float* __restrict__ zb) {  // u1,u2,p11..p22
  int idx = blockIdx.x * 256 + threadIdx.x;
  int b = idx >> 16, rem = idx & 0xFFFF;
  int y = rem >> 8, x = rem & 255;
  const float* s2p = s2 + (size_t)b * HW;
  float dx = 0.5f * (s2p[y * Wd + min(x + 1, Wd - 1)] - s2p[y * Wd + max(x - 1, 0)]);
  float dy = 0.5f * (s2p[min(y + 1, Hd - 1) * Wd + x] - s2p[max(y - 1, 0) * Wd + x]);
  fix[idx] = dx;
  fix[PLANE + idx] = dy;
  fix[2 * (size_t)PLANE + idx] = dx * dx + dy * dy + EPSF;
  fix[3 * (size_t)PLANE + idx] = s2p[y * Wd + x] - s1[idx];
  #pragma unroll
  for (int f = 0; f < 6; ++f) zb[(size_t)f * PLANE + idx] = 0.0f;
}

#define SAT(f, yy, xx) sh[f][((yy) + 1) * 18 + ((xx) + 1)]

// u-update at one point from LDS-staged fields (entry u/p)
__device__ __forceinline__ void unew(const float sh[10][324], int lx, int ly, int gx, int gy,
                                     float& u1n, float& u2n, float& rho) {
  float u1 = SAT(0, ly, lx), u2 = SAT(1, ly, lx);
  float dx = SAT(6, ly, lx), dy = SAT(7, ly, lx);
  float gr = SAT(8, ly, lx), rc = SAT(9, ly, lx);
  rho = rc + dx * u1 + dy * u2 + EPSF;
  float th = LT * gr;
  float coef;
  if (rho < -th)      coef = LT;
  else if (rho > th)  coef = -LT;
  else                coef = (gr > EPSF) ? (-rho / gr) : 0.0f;
  float v1 = coef * dx + u1;
  float v2 = coef * dy + u2;
  float p11c = SAT(2, ly, lx), p11l = SAT(2, ly, lx - 1);
  float p12c = SAT(3, ly, lx), p12u = SAT(3, ly - 1, lx);
  float p21c = SAT(4, ly, lx), p21l = SAT(4, ly, lx - 1);
  float p22c = SAT(5, ly, lx), p22u = SAT(5, ly - 1, lx);
  float d1 = (gx == 0) ? p11c : (gx == Wd - 1 ? -p11l : p11c - p11l);
  float d2 = (gy == 0) ? p12c : (gy == Hd - 1 ? -p12u : p12c - p12u);
  float d3 = (gx == 0) ? p21c : (gx == Wd - 1 ? -p21l : p21c - p21l);
  float d4 = (gy == 0) ? p22c : (gy == Hd - 1 ? -p22u : p22c - p22u);
  u1n = v1 + THETAv * (d1 + d2);
  u2n = v2 + THETAv * (d3 + d4);
}

// one fused TV-L1 iteration: async-LDS halo staging, then u-update + p-update
__global__ void __launch_bounds__(256) k_iter(const float* __restrict__ src,
                                              float* __restrict__ dst,
                                              const float* __restrict__ fix,
                                              float* __restrict__ out, int last) {
  __shared__ float sh[10][324];
  int tb = blockIdx.x;                 // NB * 256 tiles
  int b = tb >> 8;
  int ty = (tb >> 4) & 15, tx = tb & 15;
  size_t boff = (size_t)b * HW;
  const float* sp[10] = {
    src + boff,                     src + (size_t)PLANE + boff,
    src + 2 * (size_t)PLANE + boff, src + 3 * (size_t)PLANE + boff,
    src + 4 * (size_t)PLANE + boff, src + 5 * (size_t)PLANE + boff,
    fix + boff,                     fix + (size_t)PLANE + boff,
    fix + 2 * (size_t)PLANE + boff, fix + 3 * (size_t)PLANE + boff };
  int tid = threadIdx.x;
#if HAS_ASYNC_LDS
  // stage halo tiles via CDNA5 async global->LDS (ASYNCcnt), clamped addresses
  #pragma unroll
  for (int f = 0; f < 10; ++f) {
    for (int e = tid; e < 324; e += 256) {
      int ly = e / 18 - 1, lx = e % 18 - 1;
      int gy = ty * 16 + ly, gx = tx * 16 + lx;
      int cy = min(max(gy, 0), Hd - 1), cx = min(max(gx, 0), Wd - 1);
      float* gp = const_cast<float*>(sp[f] + cy * Wd + cx);
      __builtin_amdgcn_global_load_async_to_lds_b32(
          (glb_i32_t*)(int*)gp, (lds_i32_t*)(int*)&sh[f][e], 0, 0);
    }
  }
  __builtin_amdgcn_s_wait_asynccnt(0);
  __syncthreads();
  // border fix: out-of-image halo entries must be exact zeros
  if (tx == 0 || tx == 15 || ty == 0 || ty == 15) {
    #pragma unroll
    for (int f = 0; f < 10; ++f) {
      for (int e = tid; e < 324; e += 256) {
        int ly = e / 18 - 1, lx = e % 18 - 1;
        int gy = ty * 16 + ly, gx = tx * 16 + lx;
        if ((gy < 0) | (gy >= Hd) | (gx < 0) | (gx >= Wd)) sh[f][e] = 0.0f;
      }
    }
  }
  __syncthreads();
#else
  #pragma unroll
  for (int f = 0; f < 10; ++f) {
    for (int e = tid; e < 324; e += 256) {
      int ly = e / 18 - 1, lx = e % 18 - 1;
      int gy = ty * 16 + ly, gx = tx * 16 + lx;
      bool v = (gy >= 0) & (gy < Hd) & (gx >= 0) & (gx < Wd);
      int cy = min(max(gy, 0), Hd - 1), cx = min(max(gx, 0), Wd - 1);
      float val = sp[f][cy * Wd + cx];
      sh[f][e] = v ? val : 0.0f;
    }
  }
  __syncthreads();
#endif
  int lx = tid & 15, ly = tid >> 4;
  int gx = tx * 16 + lx, gy = ty * 16 + ly;
  float u1c, u2c, rho, tmp;
  unew(sh, lx, ly, gx, gy, u1c, u2c, rho);
  float u1r = u1c, u2r = u2c, u1d = u1c, u2d = u2c;
  if (gx < Wd - 1) unew(sh, lx + 1, ly, gx + 1, gy, u1r, u2r, tmp);
  if (gy < Hd - 1) unew(sh, lx, ly + 1, gx, gy + 1, u1d, u2d, tmp);
  float u1x = (gx < Wd - 1) ? u1r - u1c : 0.0f;
  float u1y = (gy < Hd - 1) ? u1d - u1c : 0.0f;
  float u2x = (gx < Wd - 1) ? u2r - u2c : 0.0f;
  float u2y = (gy < Hd - 1) ? u2d - u2c : 0.0f;
  float r1 = 1.0f / (1.0f + TAUT * sqrtf(u1x * u1x + u1y * u1y + EPSF));
  float r2 = 1.0f / (1.0f + TAUT * sqrtf(u2x * u2x + u2y * u2y + EPSF));
  int pi = gy * Wd + gx;
  dst[boff + pi]                     = u1c;
  dst[(size_t)PLANE + boff + pi]     = u2c;
  dst[2 * (size_t)PLANE + boff + pi] = (SAT(2, ly, lx) + TAUT * u1x) * r1;
  dst[3 * (size_t)PLANE + boff + pi] = (SAT(3, ly, lx) + TAUT * u1y) * r1;
  dst[4 * (size_t)PLANE + boff + pi] = (SAT(4, ly, lx) + TAUT * u2x) * r2;
  dst[5 * (size_t)PLANE + boff + pi] = (SAT(5, ly, lx) + TAUT * u2y) * r2;
  if (last) {
    float* ob = out + (size_t)b * 3 * HW;
    ob[pi] = u1c; ob[HW + pi] = u2c; ob[2 * HW + pi] = rho;
  }
}

extern "C" void kernel_launch(void* const* d_in, const int* in_sizes, int n_in,
                              void* d_out, int out_size, void* d_ws, size_t ws_size,
                              hipStream_t stream) {
  (void)in_sizes; (void)n_in; (void)out_size; (void)ws_size;
  const float* x1 = (const float*)d_in[0];
  const float* x2 = (const float*)d_in[1];
  float* ws  = (float*)d_ws;
  float* out = (float*)d_out;
  // plane layout (each PLANE floats): 0:g1 1:g2 2:t1 3:t2 4:s1 5:s2
  // fix reuses planes 0..3 after smoothing; bufA=6..11, bufB=12..17  (36MB total)
  float* g1   = ws;
  float* t1   = ws + 2 * (size_t)PLANE;
  float* s1   = ws + 4 * (size_t)PLANE;
  float* fix  = ws;                         // dx,dy,grad,rhoc overwrite g/t (dead)
  float* bufA = ws + 6 * (size_t)PLANE;
  float* bufB = ws + 12 * (size_t)PLANE;
  unsigned* mm = (unsigned*)(ws + 18 * (size_t)PLANE);

  k_init<<<1, 32, 0, stream>>>(mm);
  k_gray<<<PLANE / 256, 256, 0, stream>>>(x1, x2, g1, g1 + PLANE, mm);
  k_gauss_row<<<4096, 32, 0, stream>>>(g1, t1, mm);   // (g1||g2) -> (t1||t2)
  k_gauss_col<<<4096, 32, 0, stream>>>(t1, s1);       // (t1||t2) -> (s1||s2)
  k_grad<<<PLANE / 256, 256, 0, stream>>>(s1, s1 + PLANE, fix, bufA);
  for (int i = 0; i < 30; ++i) {
    const float* src = (i & 1) ? bufB : bufA;
    float*       dst = (i & 1) ? bufA : bufB;
    k_iter<<<NB * 256, 256, 0, stream>>>(src, dst, fix, out, i == 29 ? 1 : 0);
  }
}
